// GroupedQueryAttention_76467597738248
// MI455X (gfx1250) — compile-verified, matrix operand order checked by reading, not run on hardware
//
#include <hip/hip_runtime.h>
#include <hip/hip_bf16.h>

// ---------------------------------------------------------------------------
// GQA attention for MI455X (gfx1250): all matmuls via v_wmma_f32_16x16x32_bf16
// ---------------------------------------------------------------------------

typedef __bf16 bf16_t;
typedef __bf16 v8bf  __attribute__((ext_vector_type(8)));
typedef __bf16 v16bf __attribute__((ext_vector_type(16)));
typedef float  v8f   __attribute__((ext_vector_type(8)));

#define BATCH   2
#define SEQ     2048
#define DIM     2048
#define NHEADS  32
#define NKV     8
#define HDIM    64
#define KVDIM   (NKV * HDIM)      // 512
#define BL      (BATCH * SEQ)     // 4096

// ln(10000)/ (HDIM/2)
#define NEG_LOGBASE_OVER_HALF (-0.28782313662425572f)

// ---- WMMA helpers ----------------------------------------------------------

__device__ __forceinline__ v8f wmma_bf16(v16bf a, v16bf b, v8f c) {
  // D = A(16x32 bf16) * B(32x16 bf16) + C(16x16 f32)
  return __builtin_amdgcn_wmma_f32_16x16x32_bf16(
      /*neg_a=*/false, a, /*neg_b=*/false, b,
      /*c_mod=*/(short)0, c, /*reuse_a=*/false, /*reuse_b=*/false);
}

// Load a 16x32 A-style fragment row for this lane from a row-major bf16 matrix.
// Per ISA 7.12.2 (16-bit A 16x32): lane<16 holds K={0..7,16..23}, lane>=16
// holds K={8..15,24..31}. rowptr points at element K=0 of this lane's row.
// koff = (lane<16) ? 0 : 8.  Both halves are contiguous 16B loads.
__device__ __forceinline__ v16bf load_frag(const bf16_t* rowptr, int koff) {
  v8bf lo = *(const v8bf*)(rowptr + koff);
  v8bf hi = *(const v8bf*)(rowptr + 16 + koff);
  return __builtin_shufflevector(lo, hi, 0, 1, 2, 3, 4, 5, 6, 7,
                                          8, 9, 10, 11, 12, 13, 14, 15);
}

// ---- elementwise staging kernels ------------------------------------------

__global__ __launch_bounds__(256) void cvt_f32_bf16(const float* __restrict__ in,
                                                    bf16_t* __restrict__ out,
                                                    int n) {
  int i = blockIdx.x * 256 + threadIdx.x;
  if (i < n) out[i] = (bf16_t)in[i];
}

// W [K x N] f32 row-major  ->  Wt [N x K] bf16 row-major (coalesced writes)
__global__ __launch_bounds__(256) void transpose_cvt(const float* __restrict__ W,
                                                     bf16_t* __restrict__ Wt,
                                                     int K, int N) {
  int idx = blockIdx.x * 256 + threadIdx.x;
  if (idx >= N * K) return;
  int n = idx / K;
  int k = idx - n * K;
  Wt[idx] = (bf16_t)W[(size_t)k * N + n];
}

// ---- generic bf16 WMMA GEMM: C[MxN] = A[MxK] * Bt[NxK]^T ------------------
// Block = 256 threads = 8 waves arranged 2(M) x 4(N); wave tile = 32x64.
// Block tile = 64(M) x 256(N).  M,N,K all multiples of the tile sizes here.

__global__ __launch_bounds__(256) void gemm_bf16(const bf16_t* __restrict__ A,
                                                 const bf16_t* __restrict__ Bt,
                                                 float* __restrict__ C,
                                                 int M, int N, int K) {
  const int lane  = threadIdx.x & 31;
  const int wave  = threadIdx.x >> 5;
  const int l15   = lane & 15;
  const int koff  = (lane < 16) ? 0 : 8;
  const int chalf = (lane < 16) ? 0 : 8;

  const int m0 = blockIdx.x * 64 + (wave >> 2) * 32;
  const int n0 = blockIdx.y * 256 + (wave & 3) * 64;

  v8f acc[2][4];
#pragma unroll
  for (int mi = 0; mi < 2; ++mi)
#pragma unroll
    for (int ni = 0; ni < 4; ++ni)
#pragma unroll
      for (int r = 0; r < 8; ++r) acc[mi][ni][r] = 0.0f;

  const bf16_t* arow[2];
  const bf16_t* brow[4];
#pragma unroll
  for (int mi = 0; mi < 2; ++mi)
    arow[mi] = A + (size_t)(m0 + mi * 16 + l15) * K;
#pragma unroll
  for (int ni = 0; ni < 4; ++ni)
    brow[ni] = Bt + (size_t)(n0 + ni * 16 + l15) * K;

  for (int k = 0; k < K; k += 32) {
    v16bf af[2], bfr[4];
#pragma unroll
    for (int mi = 0; mi < 2; ++mi) af[mi] = load_frag(arow[mi] + k, koff);
#pragma unroll
    for (int ni = 0; ni < 4; ++ni) bfr[ni] = load_frag(brow[ni] + k, koff);
#pragma unroll
    for (int mi = 0; mi < 2; ++mi)
#pragma unroll
      for (int ni = 0; ni < 4; ++ni)
        acc[mi][ni] = wmma_bf16(af[mi], bfr[ni], acc[mi][ni]);
  }

#pragma unroll
  for (int mi = 0; mi < 2; ++mi)
#pragma unroll
    for (int ni = 0; ni < 4; ++ni)
#pragma unroll
      for (int r = 0; r < 8; ++r) {
        int row = m0 + mi * 16 + r + chalf;
        int col = n0 + ni * 16 + l15;
        C[(size_t)row * N + col] = acc[mi][ni][r];
      }
}

// ---- RoPE staging ----------------------------------------------------------
// Q: [BL, DIM] f32 -> head-major bf16 [B, H, L, 64] with RoPE applied.
__global__ __launch_bounds__(256) void rope_q(const float* __restrict__ Qf,
                                              bf16_t* __restrict__ Qb) {
  int idx = blockIdx.x * 256 + threadIdx.x;       // B*L*H*32 threads
  int i = idx & 31;
  int h = (idx >> 5) & (NHEADS - 1);
  int l = (idx >> 10) & (SEQ - 1);
  int b = idx >> 21;
  float invf = __expf((float)i * NEG_LOGBASE_OVER_HALF);
  float s, c;
  __sincosf((float)l * invf, &s, &c);
  size_t src = (size_t)(b * SEQ + l) * DIM + h * HDIM + i;
  float q0 = Qf[src], q1 = Qf[src + 32];
  size_t dst = ((size_t)(b * NHEADS + h) * SEQ + l) * HDIM + i;
  Qb[dst]      = (bf16_t)(q0 * c - q1 * s);
  Qb[dst + 32] = (bf16_t)(q1 * c + q0 * s);
}

// K: RoPE -> [B, G, L, 64];  V: no RoPE, transposed -> [B, G, 64, L].
__global__ __launch_bounds__(256) void rope_k_stage_v(const float* __restrict__ Kf,
                                                      const float* __restrict__ Vf,
                                                      bf16_t* __restrict__ Kb,
                                                      bf16_t* __restrict__ Vt) {
  int idx = blockIdx.x * 256 + threadIdx.x;       // B*L*G*32 threads
  int i = idx & 31;
  int g = (idx >> 5) & (NKV - 1);
  int l = (idx >> 8) & (SEQ - 1);
  int b = idx >> 19;
  float invf = __expf((float)i * NEG_LOGBASE_OVER_HALF);
  float s, c;
  __sincosf((float)l * invf, &s, &c);
  size_t src = (size_t)(b * SEQ + l) * KVDIM + g * HDIM + i;
  float k0 = Kf[src], k1 = Kf[src + 32];
  size_t kdst = ((size_t)(b * NKV + g) * SEQ + l) * HDIM + i;
  Kb[kdst]      = (bf16_t)(k0 * c - k1 * s);
  Kb[kdst + 32] = (bf16_t)(k1 * c + k0 * s);
  float v0 = Vf[src], v1 = Vf[src + 32];
  size_t vbase = (size_t)(b * NKV + g) * HDIM;
  Vt[(vbase + i) * SEQ + l]      = (bf16_t)v0;
  Vt[(vbase + i + 32) * SEQ + l] = (bf16_t)v1;
}

// ---- flash attention (causal, GQA) ----------------------------------------
// One wave handles one 16-query block of one head; keys in blocks of 32
// (matching WMMA K=32).  S = Q*K^T via 4 WMMAs, online softmax with 16-lane
// butterfly reductions, P transposed to A-layout via per-wave LDS staging,
// then O += P*V via 4 WMMAs.  Output written as [B, L, H, 64] bf16.

__global__ __launch_bounds__(256) void flash_attn(const bf16_t* __restrict__ Qb,
                                                  const bf16_t* __restrict__ Kb,
                                                  const bf16_t* __restrict__ Vt,
                                                  bf16_t* __restrict__ Ob) {
  __shared__ bf16_t lds_p[8][16 * 32];            // 1 KB per wave

  const int lane  = threadIdx.x & 31;
  const int wave  = threadIdx.x >> 5;
  const int l15   = lane & 15;
  const int koff  = (lane < 16) ? 0 : 8;
  const int chalf = (lane < 16) ? 0 : 8;

  const int nqb   = SEQ / 16;                     // 128
  int gw = blockIdx.x * 8 + wave;
  int qi = gw % nqb;  gw /= nqb;
  int h  = gw % NHEADS;
  int b  = gw / NHEADS;
  const int g     = h >> 2;                       // kv head (N_REP = 4)
  const int qbase = qi * 16;

  const bf16_t* qh = Qb + ((size_t)(b * NHEADS + h) * SEQ) * HDIM;
  const bf16_t* kh = Kb + ((size_t)(b * NKV + g) * SEQ) * HDIM;
  const bf16_t* vh = Vt + ((size_t)(b * NKV + g) * HDIM) * SEQ;
  bf16_t* myp = &lds_p[wave][0];

  // Q fragments for the two 32-deep hd steps (reused across all key blocks)
  const bf16_t* qrow = qh + (size_t)(qbase + l15) * HDIM;
  v16bf qf[2];
  qf[0] = load_frag(qrow, koff);
  qf[1] = load_frag(qrow + 32, koff);

  float m_r[8], l_r[8];
  v8f acc[4];
#pragma unroll
  for (int r = 0; r < 8; ++r) { m_r[r] = -1e30f; l_r[r] = 0.0f; }
#pragma unroll
  for (int t = 0; t < 4; ++t)
#pragma unroll
    for (int r = 0; r < 8; ++r) acc[t][r] = 0.0f;

  for (int kb = 0; kb < qbase + 16; kb += 32) {
    // ---- S tiles: keys [kb, kb+16) and [kb+16, kb+32) ----
    v8f s0, s1;
#pragma unroll
    for (int r = 0; r < 8; ++r) { s0[r] = 0.0f; s1[r] = 0.0f; }
#pragma unroll
    for (int s = 0; s < 2; ++s) {
      v16bf kf0 = load_frag(kh + (size_t)(kb + l15) * HDIM + s * 32, koff);
      v16bf kf1 = load_frag(kh + (size_t)(kb + 16 + l15) * HDIM + s * 32, koff);
      s0 = wmma_bf16(qf[s], kf0, s0);
      s1 = wmma_bf16(qf[s], kf1, s1);
    }

    // ---- online softmax over the 32 keys of this block ----
    float rowscale[8];
    const int c0 = kb + l15, c1 = c0 + 16;
#pragma unroll
    for (int r = 0; r < 8; ++r) {
      int row = qbase + r + chalf;
      float a0 = (c0 <= row) ? s0[r] * 0.125f : -1e9f;   // 1/sqrt(64)
      float a1 = (c1 <= row) ? s1[r] * 0.125f : -1e9f;
      float mb = fmaxf(a0, a1);
#pragma unroll
      for (int msk = 1; msk < 16; msk <<= 1)
        mb = fmaxf(mb, __shfl_xor(mb, msk, 32));
      float mn = fmaxf(m_r[r], mb);
      float sc = __expf(m_r[r] - mn);
      float p0 = __expf(a0 - mn);
      float p1 = __expf(a1 - mn);
      float rs = p0 + p1;
#pragma unroll
      for (int msk = 1; msk < 16; msk <<= 1)
        rs += __shfl_xor(rs, msk, 32);
      l_r[r] = l_r[r] * sc + rs;
      m_r[r] = mn;
      rowscale[r] = sc;
      int lrow = r + chalf;
      myp[lrow * 32 + l15]      = (bf16_t)p0;
      myp[lrow * 32 + 16 + l15] = (bf16_t)p1;
    }
#pragma unroll
    for (int t = 0; t < 4; ++t)
#pragma unroll
      for (int r = 0; r < 8; ++r) acc[t][r] *= rowscale[r];

    // ---- O += P * V  (P: 16x32 from LDS in A layout; same-wave DS in-order)
    v16bf pf = load_frag(myp + l15 * 32, koff);
#pragma unroll
    for (int t = 0; t < 4; ++t) {
      v16bf vf = load_frag(vh + (size_t)(t * 16 + l15) * SEQ + kb, koff);
      acc[t] = wmma_bf16(pf, vf, acc[t]);
    }
  }

  // ---- normalize and write out as [B, L, H, 64] bf16 ----
#pragma unroll
  for (int t = 0; t < 4; ++t)
#pragma unroll
    for (int r = 0; r < 8; ++r) {
      int row = qbase + r + chalf;
      int col = t * 16 + l15;
      float o = acc[t][r] / l_r[r];
      Ob[((size_t)(b * SEQ + row) * NHEADS + h) * HDIM + col] = (bf16_t)o;
    }
}

// ---------------------------------------------------------------------------

static inline size_t align256(size_t x) { return (x + 255) & ~(size_t)255; }

extern "C" void kernel_launch(void* const* d_in, const int* in_sizes, int n_in,
                              void* d_out, int out_size, void* d_ws, size_t ws_size,
                              hipStream_t stream) {
  const float* x  = (const float*)d_in[0];
  const float* wq = (const float*)d_in[1];
  const float* wk = (const float*)d_in[2];
  const float* wv = (const float*)d_in[3];
  const float* wo = (const float*)d_in[4];

  // workspace layout (~108 MB total)
  char* ws = (char*)d_ws;
  size_t off = 0;
  auto take = [&](size_t bytes) { char* p = ws + off; off = align256(off + bytes); return p; };
  bf16_t* xb   = (bf16_t*)take((size_t)BL * DIM * 2);
  bf16_t* wqt  = (bf16_t*)take((size_t)DIM * DIM * 2);
  bf16_t* wkt  = (bf16_t*)take((size_t)KVDIM * DIM * 2);
  bf16_t* wvt  = (bf16_t*)take((size_t)KVDIM * DIM * 2);
  bf16_t* wot  = (bf16_t*)take((size_t)DIM * DIM * 2);
  float*  Qf   = (float*)take((size_t)BL * DIM * 4);
  float*  Kf   = (float*)take((size_t)BL * KVDIM * 4);
  float*  Vf   = (float*)take((size_t)BL * KVDIM * 4);
  bf16_t* Qbh  = (bf16_t*)take((size_t)BL * DIM * 2);
  bf16_t* Kbh  = (bf16_t*)take((size_t)BL * KVDIM * 2);
  bf16_t* Vth  = (bf16_t*)take((size_t)BL * KVDIM * 2);
  bf16_t* attnb = (bf16_t*)Qf;   // Qf is dead after rope_q; reuse (32MB >= 16MB)

  // 1) stage inputs to bf16 (weights transposed to [N][K] for B-fragments)
  cvt_f32_bf16<<<(BL * DIM + 255) / 256, 256, 0, stream>>>(x, xb, BL * DIM);
  transpose_cvt<<<((size_t)DIM * DIM + 255) / 256, 256, 0, stream>>>(wq, wqt, DIM, DIM);
  transpose_cvt<<<((size_t)DIM * KVDIM + 255) / 256, 256, 0, stream>>>(wk, wkt, DIM, KVDIM);
  transpose_cvt<<<((size_t)DIM * KVDIM + 255) / 256, 256, 0, stream>>>(wv, wvt, DIM, KVDIM);
  transpose_cvt<<<((size_t)DIM * DIM + 255) / 256, 256, 0, stream>>>(wo, wot, DIM, DIM);

  // 2) projections (WMMA GEMMs, f32 out)
  gemm_bf16<<<dim3(BL / 64, DIM / 256), 256, 0, stream>>>(xb, wqt, Qf, BL, DIM, DIM);
  gemm_bf16<<<dim3(BL / 64, KVDIM / 256), 256, 0, stream>>>(xb, wkt, Kf, BL, KVDIM, DIM);
  gemm_bf16<<<dim3(BL / 64, KVDIM / 256), 256, 0, stream>>>(xb, wvt, Vf, BL, KVDIM, DIM);

  // 3) RoPE + head-major bf16 staging (V transposed for PV B-fragments)
  rope_q<<<(BATCH * SEQ * NHEADS * 32) / 256, 256, 0, stream>>>(Qf, Qbh);
  rope_k_stage_v<<<(BATCH * SEQ * NKV * 32) / 256, 256, 0, stream>>>(Kf, Vf, Kbh, Vth);

  // 4) causal GQA flash attention (WMMA), output [B,L,H,64] bf16
  flash_attn<<<(BATCH * NHEADS * (SEQ / 16)) / 8, 256, 0, stream>>>(Qbh, Kbh, Vth, attnb);

  // 5) output projection -> f32 d_out
  gemm_bf16<<<dim3(BL / 64, DIM / 256), 256, 0, stream>>>(attnb, wot, (float*)d_out,
                                                          BL, DIM, DIM);
}
